// SingleAttentionHead_35201551958336
// MI455X (gfx1250) — matrix operand order repaired and checked by reading, not compiled, not run on hardware
//
#include <hip/hip_runtime.h>

// CDNA5 / gfx1250, wave32. WMMA f32_16x16x32_f16 + async global->LDS staging.

typedef __attribute__((ext_vector_type(16))) _Float16 v16h;
typedef __attribute__((ext_vector_type(8)))  float    v8f;
typedef int v4i __attribute__((vector_size(16)));   // matches builtin proto

#define EMB   1024
#define HEAD  64
#define BATCH 8
#define SEQ   2048
#define KBLK  32

static __device__ __forceinline__ float fast_exp2(float x) {
#if __has_builtin(__builtin_amdgcn_exp2f)
  return __builtin_amdgcn_exp2f(x);
#else
  return exp2f(x);
#endif
}

// ---- async global->LDS copy of 16 bytes per lane (guarded; safe fallback) ----
#if __has_builtin(__builtin_amdgcn_global_load_async_to_lds_b128) && \
    __has_builtin(__builtin_amdgcn_s_wait_asynccnt)
#define USE_ASYNC_LDS 1
#endif

static __device__ __forceinline__ void async_copy16(const _Float16* src,
                                                    _Float16* dst_lds) {
#ifdef USE_ASYNC_LDS
  __builtin_amdgcn_global_load_async_to_lds_b128(
      (__attribute__((address_space(1))) v4i*)src,
      (__attribute__((address_space(3))) v4i*)dst_lds, 0, 0);
#else
  *(uint4*)dst_lds = *(const uint4*)src;
#endif
}

static __device__ __forceinline__ void wait_async_done() {
#ifdef USE_ASYNC_LDS
  __builtin_amdgcn_s_wait_asynccnt(0);
#endif
}

// ---------------------------------------------------------------------------
// Phase 0: convert Wq|Wk|Wv (f32 [64,1024] each) into packed fp16 Wh[3][64][1024].
// ---------------------------------------------------------------------------
__global__ void __launch_bounds__(256)
wcvt_kernel(const float* __restrict__ Wq, const float* __restrict__ Wk,
            const float* __restrict__ Wv, _Float16* __restrict__ Wh)
{
  const int i = blockIdx.x * 256 + threadIdx.x;     // 0 .. 3*65536-1
  const int per = HEAD * EMB;
  const float* src = (i < per) ? Wq : (i < 2 * per) ? Wk : Wv;
  Wh[i] = (_Float16)src[i & (per - 1)];
}

// ---------------------------------------------------------------------------
// Phase 1: fused q/k/v projection — ONE pass over x. 8 waves per block share a
// 128-row x band; the per-step W slab (192 rows x 32 cols fp16 = 12 KB) is
// staged cooperatively into double-buffered LDS with async global->LDS, so the
// B operands come from ds_load instead of waited global loads, and each block
// reads W from L2 once instead of 8x. Uniform 32-step loop -> barriers legal.
// ---------------------------------------------------------------------------
__global__ void __launch_bounds__(256)
qkv_proj_kernel(const float* __restrict__ x,
                const _Float16* __restrict__ Wh,   // [3*64][1024] fp16
                _Float16* __restrict__ Qh,         // [B*T, 64], scaled by 1/8
                _Float16* __restrict__ Kh,         // [B*T, 64]
                _Float16* __restrict__ Vt)         // [B, 64, T]
{
  __shared__ _Float16 wbuf[2][192 * 32];           // 12 KB per buffer

  const int tid  = threadIdx.x;
  const int lane = tid & 31;
  const int wave = tid >> 5;
  const int tile = blockIdx.x * 8 + wave;
  const int row0 = tile * 16;

  const int n   = lane & 15;
  const int hi  = lane >> 4;
  const int kbA = hi ? 16 : 0;
  const int kbB = hi ? 16 : 0;

  v8f acc[3][4] = {};
  const float* xrow = x + (size_t)(row0 + n) * EMB;

  // Cooperative staging: 768 x 16B chunks per step, 3 per thread.
  // chunk i: W row = i>>2 (0..191), half-offset = (i&3)*8.
  // Prime buffer 0 with step 0 (cols 0..31).
#pragma unroll
  for (int q = 0; q < 3; ++q) {
    const int i  = tid * 3 + q;
    const int wr = i >> 2, wo = (i & 3) * 8;
    async_copy16(Wh + (size_t)wr * EMB + wo, &wbuf[0][wr * 32 + wo]);
  }
  wait_async_done();
  __syncthreads();

  for (int s = 0; s < 32; ++s) {
    const int cur = s & 1;
    const int c0  = s * 32;

    // Prefetch next W slab into the other buffer (overlapped with compute).
    if (s < 31) {
#pragma unroll
      for (int q = 0; q < 3; ++q) {
        const int i  = tid * 3 + q;
        const int wr = i >> 2, wo = (i & 3) * 8;
        async_copy16(Wh + (size_t)wr * EMB + c0 + 32 + wo,
                     &wbuf[cur ^ 1][wr * 32 + wo]);
      }
    }

    // A operand: 16-row x tile, fp32 -> fp16 (each x element read once).
    v16h a;
#pragma unroll
    for (int e = 0; e < 8; ++e) a[e]     = (_Float16)xrow[c0 + kbA + e];
#pragma unroll
    for (int e = 0; e < 8; ++e) a[8 + e] = (_Float16)xrow[c0 + 8 + kbA + e];

    // 12 WMMAs, B operands from LDS.
#pragma unroll
    for (int w = 0; w < 3; ++w) {
#pragma unroll
      for (int hc = 0; hc < 4; ++hc) {
        const _Float16* wrp = &wbuf[cur][(w * 64 + hc * 16 + n) * 32 + kbB];
        v16h bm = *(const v16h*)wrp;
        acc[w][hc] = __builtin_amdgcn_wmma_f32_16x16x32_f16(
            false, a, false, bm, (short)0, acc[w][hc], false, false);
      }
    }

    wait_async_done();     // next slab fully landed in LDS
    __syncthreads();       // all waves done reading current slab
  }

  // q (scaled by 1/sqrt(64)) and k: row-major [B*T, 64]
#pragma unroll
  for (int hc = 0; hc < 4; ++hc) {
#pragma unroll
    for (int r = 0; r < 8; ++r) {
      const int row = row0 + r + hi * 8;
      Qh[(size_t)row * HEAD + hc * 16 + n] = (_Float16)(acc[0][hc][r] * 0.125f);
      Kh[(size_t)row * HEAD + hc * 16 + n] = (_Float16)acc[1][hc][r];
    }
  }
  // v transposed: Vt[b][h][t]
#pragma unroll
  for (int hc = 0; hc < 4; ++hc) {
#pragma unroll
    for (int r = 0; r < 8; ++r) {
      const int row = row0 + r + hi * 8;
      const int b = row >> 11;
      const int t = row & (SEQ - 1);
      Vt[((size_t)b * HEAD + hc * 16 + n) * SEQ + t] = (_Float16)acc[2][hc][r];
    }
  }
}

// ---------------------------------------------------------------------------
// Phase 2: block-cooperative causal flash attention.
// 8 waves share a 128-row band; key blocks of 32 staged into double-buffered
// LDS via async global->LDS; lockstep iteration; waves above the diagonal skip
// compute. Per block-iter per wave: 4 WMMAs for S, online softmax, LDS C->A
// transpose of P, 4 WMMAs for P@V.
// ---------------------------------------------------------------------------
__global__ void __launch_bounds__(256)
attn_kernel(const _Float16* __restrict__ Qh,
            const _Float16* __restrict__ Kh,
            const _Float16* __restrict__ Vt,
            float* __restrict__ out)
{
  __shared__ _Float16 kbuf[2][KBLK * HEAD];   // 4 KB per buffer
  __shared__ _Float16 vbuf[2][HEAD * KBLK];   // 4 KB per buffer
  __shared__ _Float16 pshare[8][16 * 32];     // 1 KB per wave

  const int tid  = threadIdx.x;
  const int lane = tid & 31;
  const int wave = tid >> 5;

  const int band = blockIdx.x;                // 128 bands of 128 rows
  const int b    = band >> 4;
  const int tb0  = (band & 15) * 128;
  const int t0   = tb0 + wave * 16;

  const int n   = lane & 15;
  const int hi  = lane >> 4;
  const int kbA = hi ? 16 : 0;
  const int kbB = hi ? 16 : 0;

  const _Float16* Qb = Qh + (size_t)b * SEQ * HEAD;
  const _Float16* Kb = Kh + (size_t)b * SEQ * HEAD;
  const _Float16* Vb = Vt + (size_t)b * HEAD * SEQ;

  const int krow = tid >> 3, koff = (tid & 7) * 8;     // K: 32 rows x 64 halves
  const int vrow = tid >> 2, voff = (tid & 3) * 8;     // V: 64 rows x 32 halves

  // Q tile in A layout (loop-invariant).
  v16h qa[2];
  {
    const _Float16* qrow = Qb + (size_t)(t0 + n) * HEAD;
#pragma unroll
    for (int c = 0; c < 2; ++c) {
#pragma unroll
      for (int e = 0; e < 8; ++e) qa[c][e]     = qrow[c * 32 + kbA + e];
#pragma unroll
      for (int e = 0; e < 8; ++e) qa[c][8 + e] = qrow[c * 32 + 8 + kbA + e];
    }
  }

  v8f oacc[4] = {};
  float mrow[8], lrow[8];
#pragma unroll
  for (int r = 0; r < 8; ++r) { mrow[r] = -1e30f; lrow[r] = 1e-30f; }

  const int jmax = (tb0 + 127) >> 5;          // uniform across the block

  async_copy16(Kb + (size_t)krow * HEAD + koff, &kbuf[0][krow * HEAD + koff]);
  async_copy16(Vb + (size_t)vrow * SEQ + voff, &vbuf[0][vrow * KBLK + voff]);
  wait_async_done();
  __syncthreads();

  for (int j = 0; j <= jmax; ++j) {
    const int cur  = j & 1;
    const int key0 = j * KBLK;

    if (j < jmax) {
      const int nk0 = key0 + KBLK;
      async_copy16(Kb + (size_t)(nk0 + krow) * HEAD + koff,
                   &kbuf[cur ^ 1][krow * HEAD + koff]);
      async_copy16(Vb + (size_t)vrow * SEQ + nk0 + voff,
                   &vbuf[cur ^ 1][vrow * KBLK + voff]);
    }

    if (key0 <= t0 + 15) {
      // ---- S = q @ k^T from LDS ----
      v8f s[2] = {};
#pragma unroll
      for (int nt = 0; nt < 2; ++nt) {
#pragma unroll
        for (int c = 0; c < 2; ++c) {
          const _Float16* kr =
              &kbuf[cur][(nt * 16 + n) * HEAD + c * 32 + kbB];
          v16h kmat = *(const v16h*)kr;
          s[nt] = __builtin_amdgcn_wmma_f32_16x16x32_f16(
              false, qa[c], false, kmat, (short)0, s[nt], false, false);
        }
      }

      // ---- causal mask on diagonal block ----
      if (key0 + KBLK - 1 > t0) {
#pragma unroll
        for (int nt = 0; nt < 2; ++nt) {
          const int key = key0 + nt * 16 + n;
#pragma unroll
          for (int r = 0; r < 8; ++r) {
            const int t = t0 + r + hi * 8;
            if (key > t) s[nt][r] = -1e30f;
          }
        }
      }

      // ---- online softmax (16-lane xor reductions) ----
      float bm[8];
#pragma unroll
      for (int r = 0; r < 8; ++r) bm[r] = fmaxf(s[0][r], s[1][r]);
#pragma unroll
      for (int off = 1; off < 16; off <<= 1) {
#pragma unroll
        for (int r = 0; r < 8; ++r)
          bm[r] = fmaxf(bm[r], __shfl_xor(bm[r], off, 32));
      }

      float alpha[8], p0[8], p1[8];
#pragma unroll
      for (int r = 0; r < 8; ++r) {
        const float mnew = fmaxf(mrow[r], bm[r]);
        alpha[r] = fast_exp2((mrow[r] - mnew) * 1.44269504f);
        p0[r]    = fast_exp2((s[0][r] - mnew) * 1.44269504f);
        p1[r]    = fast_exp2((s[1][r] - mnew) * 1.44269504f);
        mrow[r]  = mnew;
      }

      float rs[8];
#pragma unroll
      for (int r = 0; r < 8; ++r) rs[r] = p0[r] + p1[r];
#pragma unroll
      for (int off = 1; off < 16; off <<= 1) {
#pragma unroll
        for (int r = 0; r < 8; ++r)
          rs[r] += __shfl_xor(rs[r], off, 32);
      }
#pragma unroll
      for (int r = 0; r < 8; ++r) {
        lrow[r] = lrow[r] * alpha[r] + rs[r];
#pragma unroll
        for (int hc = 0; hc < 4; ++hc) oacc[hc][r] *= alpha[r];
      }

      // ---- transpose P (C-layout -> A-layout) via per-wave LDS scratch ----
      _Float16* ps = &pshare[wave][0];
#pragma unroll
      for (int r = 0; r < 8; ++r) {
        const int row = r + hi * 8;
        ps[row * 32 + n]      = (_Float16)p0[r];
        ps[row * 32 + 16 + n] = (_Float16)p1[r];
      }
      v16h pa;
      {
        const _Float16* prow = ps + n * 32;
#pragma unroll
        for (int e = 0; e < 8; ++e) pa[e]     = prow[kbA + e];
#pragma unroll
        for (int e = 0; e < 8; ++e) pa[8 + e] = prow[8 + kbA + e];
      }

      // ---- O += P @ V from LDS ----
#pragma unroll
      for (int hc = 0; hc < 4; ++hc) {
        const _Float16* vr = &vbuf[cur][(hc * 16 + n) * KBLK + kbB];
        v16h vmat = *(const v16h*)vr;
        oacc[hc] = __builtin_amdgcn_wmma_f32_16x16x32_f16(
            false, pa, false, vmat, (short)0, oacc[hc], false, false);
      }
    }

    wait_async_done();
    __syncthreads();
  }

  // ---- normalize and store ----
#pragma unroll
  for (int hc = 0; hc < 4; ++hc) {
#pragma unroll
    for (int r = 0; r < 8; ++r) {
      const int t = t0 + r + hi * 8;
      out[((size_t)b * SEQ + t) * HEAD + hc * 16 + n] = oacc[hc][r] / lrow[r];
    }
  }
}

// ---------------------------------------------------------------------------
extern "C" void kernel_launch(void* const* d_in, const int* in_sizes, int n_in,
                              void* d_out, int out_size, void* d_ws, size_t ws_size,
                              hipStream_t stream)
{
  const float* x  = (const float*)d_in[0];
  const float* Wq = (const float*)d_in[1];
  const float* Wk = (const float*)d_in[2];
  const float* Wv = (const float*)d_in[3];
  float* out = (float*)d_out;

  _Float16* Qh = (_Float16*)d_ws;
  _Float16* Kh = Qh + (size_t)BATCH * SEQ * HEAD;
  _Float16* Vt = Kh + (size_t)BATCH * SEQ * HEAD;
  _Float16* Wh = Vt + (size_t)BATCH * SEQ * HEAD;

  wcvt_kernel<<<3 * HEAD * EMB / 256, 256, 0, stream>>>(Wq, Wk, Wv, Wh);

  qkv_proj_kernel<<<(BATCH * SEQ / 16) / 8, 256, 0, stream>>>(x, Wh, Qh, Kh, Vt);

  attn_kernel<<<(BATCH * SEQ) / 128, 256, 0, stream>>>(Qh, Kh, Vt, out);
}
